// QuantAttentionFused_44942537786217
// MI455X (gfx1250) — matrix operand order repaired and checked by reading, not compiled, not run on hardware
//
#include <hip/hip_runtime.h>
#include <hip/hip_bf16.h>

// ---------------------------------------------------------------------------
// Fused attention layer for gfx1250 (MI455X): fp32 in/out, bf16 WMMA compute.
//   hidden(1,2048,4096) @ Wqkv(6144,4096)^T -> qkv(2048,6144)   [GEMM, WMMA]
//   RoPE on q/k slices of qkv                                    [elementwise]
//   causal GQA flash attention per head                          [WMMA]
//   out(2048,4096) @ Wo(4096,4096)^T -> d_out                    [GEMM, WMMA]
// ---------------------------------------------------------------------------

typedef __attribute__((ext_vector_type(16))) __bf16 v16bf;
typedef __attribute__((ext_vector_type(8)))  __bf16 v8bf;
typedef __attribute__((ext_vector_type(8)))  float  v8f;

#define S_LEN    2048
#define HID      4096
#define NQH      32
#define NKV      8
#define HD       128
#define QKV_COLS 6144   // (32 + 2*8) * 128

// fp32 -> bf16 round-to-nearest-even
__device__ __forceinline__ unsigned int f2bf(float f) {
  unsigned int u = __float_as_uint(f);
  u += 0x7fffu + ((u >> 16) & 1u);
  return u >> 16;
}
// two fp32 -> packed bf16x2 in a u32
__device__ __forceinline__ unsigned int f2bf2(float a, float b) {
  return f2bf(a) | (f2bf(b) << 16);
}

// A-fragment (16x32 bf16, MxK), ISA 7.12.2 layout:
//  lane<16: row M=lane, K={0..7}U{16..23};  lane>=16: row M=lane-16, K={8..15}U{24..31}
__device__ __forceinline__ v16bf load_frag_a(const __bf16* base, int ldk, int lane) {
  const int m  = lane & 15;
  const int kb = (lane >> 4) ? 8 : 0;
  const __bf16* p = base + m * ldk + kb;
  v8bf lo = *(const v8bf*)(p);        // K = kb .. kb+7
  v8bf hi = *(const v8bf*)(p + 16);   // K = kb+16 .. kb+23
  v16bf r;
#pragma unroll
  for (int i = 0; i < 8; ++i) { r[i] = lo[i]; r[8 + i] = hi[i]; }
  return r;
}

// B-fragment (32x16 bf16, KxN), stored in LDS as [N][K] (B^T row-major):
//  lane<16: col N=lane, K={0..7}U{16..23};  lane>=16: col N=lane-16, K={8..15}U{24..31}
__device__ __forceinline__ v16bf load_frag_b(const __bf16* base, int ldk, int lane) {
  const int n  = lane & 15;
  const int kb = (lane >> 4) ? 8 : 0;
  const __bf16* p = base + n * ldk + kb;
  v8bf lo = *(const v8bf*)(p);
  v8bf hi = *(const v8bf*)(p + 16);
  v16bf r;
#pragma unroll
  for (int i = 0; i < 8; ++i) { r[i] = lo[i]; r[8 + i] = hi[i]; }
  return r;
}

__device__ __forceinline__ v8f wmma_bf16(v16bf a, v16bf b, v8f c) {
  return __builtin_amdgcn_wmma_f32_16x16x32_bf16(false, a, false, b, (short)0, c,
                                                 false, false);
}

// ---------------------------------------------------------------------------
// GEMM: C[M,N] = A[M,K] * B[N,K]^T   (fp32 in, bf16 WMMA, fp32 out)
// block = 256 threads = 8 waves (2x4), block tile 64x256, wave tile 32x64,
// K-step 32. Requires M%64==0, N%256==0, K%32==0 (true for all our shapes).
// Staging: global float4 loads -> packed bf16 -> ds_store_b64.
// ---------------------------------------------------------------------------
__global__ __launch_bounds__(256)
void gemm_bf16_nt(const float* __restrict__ A, const float* __restrict__ B,
                  float* __restrict__ C, int M, int N, int K) {
  __shared__ __attribute__((aligned(16))) unsigned short As[64 * 32];
  __shared__ __attribute__((aligned(16))) unsigned short Bs[256 * 32];

  const int tid  = threadIdx.x;
  const int lane = tid & 31;
  const int wid  = tid >> 5;                 // 0..7
  const int bm   = blockIdx.y * 64;
  const int bn   = blockIdx.x * 256;
  const int wm   = (wid >> 2) * 32;          // 0 or 32
  const int wn   = (wid & 3) * 64;           // 0..192

  v8f acc[2][4] = {};

  for (int k0 = 0; k0 < K; k0 += 32) {
    __syncthreads();
    if (k0 + 32 < K) {                       // prefetch next K tile
      __builtin_prefetch(A + (size_t)(bm + (tid >> 2)) * K + k0 + 32, 0, 0);
      __builtin_prefetch(B + (size_t)(bn + tid) * K + k0 + 32, 0, 0);
    }
    // A tile: 64x32 = 512 float4s, 256 threads -> 2 each
#pragma unroll
    for (int i = 0; i < 2; ++i) {
      int v  = tid + i * 256;
      int m  = v >> 3;                       // 8 float4 per 32-wide row
      int kk = (v & 7) * 4;
      float4 f = *(const float4*)(A + (size_t)(bm + m) * K + k0 + kk);
      uint2 p; p.x = f2bf2(f.x, f.y); p.y = f2bf2(f.z, f.w);
      *(uint2*)(&As[m * 32 + kk]) = p;
    }
    // B tile: 256x32 = 2048 float4s, 256 threads -> 8 each
#pragma unroll
    for (int i = 0; i < 8; ++i) {
      int v  = tid + i * 256;
      int n  = v >> 3;
      int kk = (v & 7) * 4;
      float4 f = *(const float4*)(B + (size_t)(bn + n) * K + k0 + kk);
      uint2 p; p.x = f2bf2(f.x, f.y); p.y = f2bf2(f.z, f.w);
      *(uint2*)(&Bs[n * 32 + kk]) = p;
    }
    __syncthreads();

    const __bf16* AsB = (const __bf16*)As;
    const __bf16* BsB = (const __bf16*)Bs;
    v16bf a0 = load_frag_a(AsB + (wm + 0)  * 32, 32, lane);
    v16bf a1 = load_frag_a(AsB + (wm + 16) * 32, 32, lane);
#pragma unroll
    for (int j = 0; j < 4; ++j) {
      v16bf b = load_frag_b(BsB + (wn + j * 16) * 32, 32, lane);
      acc[0][j] = wmma_bf16(a0, b, acc[0][j]);
      acc[1][j] = wmma_bf16(a1, b, acc[1][j]);
    }
  }

  // C layout: M = vgpr + 8*(lane>=16), N = lane&15
  const int mbase = bm + wm + ((lane >> 4) << 3);
  const int nbase = bn + wn + (lane & 15);
#pragma unroll
  for (int i = 0; i < 2; ++i)
#pragma unroll
    for (int j = 0; j < 4; ++j)
#pragma unroll
      for (int r = 0; r < 8; ++r)
        C[(size_t)(mbase + i * 16 + r) * N + nbase + j * 16] = acc[i][j][r];
}

// ---------------------------------------------------------------------------
// RoPE in place on q (cols 0..4095) and k (cols 4096..5119) of qkv.
// heads 0..31 = q, 32..39 = k (k offset 4096 == 32*128, so col = head*128).
// ---------------------------------------------------------------------------
__global__ __launch_bounds__(256)
void rope_kernel(float* __restrict__ qkv) {
  int idx = blockIdx.x * blockDim.x + threadIdx.x;   // S * 40 * 64 threads
  int d    = idx & 63;
  int t    = idx >> 6;
  int head = t % (NQH + NKV);
  int s    = t / (NQH + NKV);
  if (s >= S_LEN) return;
  float* row = qkv + (size_t)s * QKV_COLS + head * HD;
  float inv_freq = __powf(10000.0f, -(float)d * (1.0f / 64.0f));
  float fr = (float)s * inv_freq;
  float c, sn;
  __sincosf(fr, &sn, &c);
  float x1 = row[d], x2 = row[d + 64];
  row[d]      = x1 * c - x2 * sn;
  row[d + 64] = x1 * sn + x2 * c;
}

// ---------------------------------------------------------------------------
// Flash attention, causal, GQA (4 q-heads per kv-head).
// grid = (S/64 q-blocks, 32 heads); block = 128 threads = 4 waves.
// Each wave owns 16 q rows; per-wave accumulators cover all 128 head dims.
// Key blocks of 32; online softmax state (m, l) in LDS.
// ---------------------------------------------------------------------------
__global__ __launch_bounds__(128)
void attn_fa(const float* __restrict__ qkv, float* __restrict__ out) {
  __shared__ __attribute__((aligned(16))) unsigned short Qs[64 * 128]; // [m][d] bf16
  __shared__ __attribute__((aligned(16))) unsigned short Ks[32 * 128]; // [key][d] bf16
  __shared__ __attribute__((aligned(16))) unsigned short Vt[128 * 32]; // [d][key] bf16
  __shared__ __attribute__((aligned(16))) float          Sc[64 * 32];  // scores fp32
  __shared__ __attribute__((aligned(16))) unsigned short Pb[64 * 32];  // probs bf16
  __shared__ float mrow[64];
  __shared__ float lrow[64];
  __shared__ float srow[64];

  const int tid  = threadIdx.x;
  const int lane = tid & 31;
  const int wid  = tid >> 5;      // 0..3 -> q rows [wid*16, wid*16+16)
  const int qb   = blockIdx.x;    // q rows qb*64 ..
  const int h    = blockIdx.y;    // q head
  const int kvh  = h >> 2;        // kv head (groups = 4)

  const size_t qcol = (size_t)h * HD;
  const size_t kcol = (size_t)HID + (size_t)kvh * HD;
  const size_t vcol = (size_t)(HID + NKV * HD) + (size_t)kvh * HD;

  // Q tile: 64x128 = 2048 float4s, 128 threads -> 16 each
  for (int i = tid; i < 64 * 32; i += 128) {
    int m = i >> 5, d = (i & 31) * 4;
    float4 f = *(const float4*)(qkv + (size_t)(qb * 64 + m) * QKV_COLS + qcol + d);
    uint2 p; p.x = f2bf2(f.x, f.y); p.y = f2bf2(f.z, f.w);
    *(uint2*)(&Qs[m * 128 + d]) = p;
  }
  if (tid < 64) { mrow[tid] = -3.0e38f; lrow[tid] = 0.0f; }

  v8f acc[8];
#pragma unroll
  for (int n = 0; n < 8; ++n) acc[n] = {};

  const float scale = 0.08838834764831845f;   // 1/sqrt(128)
  const __bf16* QsB = (const __bf16*)Qs;
  const __bf16* KsB = (const __bf16*)Ks;
  const __bf16* VtB = (const __bf16*)Vt;
  const __bf16* PbB = (const __bf16*)Pb;

  const int nkb = 2 * qb + 2;                 // causal: keys up to qb*64+63
  for (int kb = 0; kb < nkb; ++kb) {
    __syncthreads();                          // LDS reuse from previous iter
    // K tile (32x128, vectorized) and V^T tile (transposed scalar stores)
    for (int i = tid; i < 32 * 32; i += 128) {
      int ky = i >> 5, d = (i & 31) * 4;
      size_t row = (size_t)(kb * 32 + ky) * QKV_COLS;
      float4 fk = *(const float4*)(qkv + row + kcol + d);
      uint2 p; p.x = f2bf2(fk.x, fk.y); p.y = f2bf2(fk.z, fk.w);
      *(uint2*)(&Ks[ky * 128 + d]) = p;
      float4 fv = *(const float4*)(qkv + row + vcol + d);
      Vt[(d + 0) * 32 + ky] = (unsigned short)f2bf(fv.x);
      Vt[(d + 1) * 32 + ky] = (unsigned short)f2bf(fv.y);
      Vt[(d + 2) * 32 + ky] = (unsigned short)f2bf(fv.z);
      Vt[(d + 3) * 32 + ky] = (unsigned short)f2bf(fv.w);
    }
    __syncthreads();

    // S = Q * K^T for this wave's 16 rows x 32 keys
#pragma unroll
    for (int j = 0; j < 2; ++j) {
      v8f s = {};
#pragma unroll
      for (int kk = 0; kk < 4; ++kk) {        // head dim 128 = 4 * 32
        v16bf a = load_frag_a(QsB + (wid * 16) * 128 + kk * 32, 128, lane);
        v16bf b = load_frag_b(KsB + (j * 16) * 128 + kk * 32, 128, lane);
        s = wmma_bf16(a, b, s);
      }
      int nloc = j * 16 + (lane & 15);
      int key  = kb * 32 + nloc;
#pragma unroll
      for (int r = 0; r < 8; ++r) {
        int mloc = wid * 16 + r + ((lane >> 4) << 3);
        int q    = qb * 64 + mloc;
        float v  = s[r] * scale;
        if (key > q) v = -1.0e30f;            // causal mask
        Sc[mloc * 32 + nloc] = v;
      }
    }
    __syncthreads();

    // online softmax, one thread per row
    if (tid < 64) {
      int r = tid;
      float mold = mrow[r];
      float mmax = mold;
      for (int k = 0; k < 32; ++k) mmax = fmaxf(mmax, Sc[r * 32 + k]);
      float sum = 0.0f;
      for (int k = 0; k < 32; ++k) {
        float p = __expf(Sc[r * 32 + k] - mmax);
        sum += p;
        Pb[r * 32 + k] = (unsigned short)f2bf(p);
      }
      float sc = __expf(mold - mmax);
      srow[r] = sc;
      mrow[r] = mmax;
      lrow[r] = lrow[r] * sc + sum;
    }
    __syncthreads();

    // O = O * diag(srow) + P * V
    v16bf pa = load_frag_a(PbB + (wid * 16) * 32, 32, lane);
#pragma unroll
    for (int n = 0; n < 8; ++n) {
#pragma unroll
      for (int r = 0; r < 8; ++r) {
        int mloc = wid * 16 + r + ((lane >> 4) << 3);
        acc[n][r] *= srow[mloc];
      }
      v16bf b = load_frag_b(VtB + (n * 16) * 32, 32, lane);
      acc[n] = wmma_bf16(pa, b, acc[n]);
    }
  }
  __syncthreads();

  // normalize by l and write out[s][h*128 + d]
#pragma unroll
  for (int n = 0; n < 8; ++n) {
#pragma unroll
    for (int r = 0; r < 8; ++r) {
      int mloc = wid * 16 + r + ((lane >> 4) << 3);
      int q    = qb * 64 + mloc;
      int d    = n * 16 + (lane & 15);
      out[(size_t)q * (NQH * HD) + h * HD + d] = acc[n][r] / lrow[mloc];
    }
  }
}

// ---------------------------------------------------------------------------
extern "C" void kernel_launch(void* const* d_in, const int* in_sizes, int n_in,
                              void* d_out, int out_size, void* d_ws, size_t ws_size,
                              hipStream_t stream) {
  (void)in_sizes; (void)n_in; (void)out_size; (void)ws_size;
  const float* hs   = (const float*)d_in[0];   // (1, 2048, 4096)
  const float* Wqkv = (const float*)d_in[1];   // (6144, 4096)
  const float* Wo   = (const float*)d_in[2];   // (4096, 4096)
  float* outp = (float*)d_out;                 // (1, 2048, 4096)

  float* qkv  = (float*)d_ws;                          // 2048 * 6144 fp32 (48 MB)
  float* attn = qkv + (size_t)S_LEN * QKV_COLS;        // 2048 * 4096 fp32 (32 MB)

  // 1) QKV projection: qkv[s,o] = sum_h hs[s,h] * Wqkv[o,h]
  {
    dim3 grid(QKV_COLS / 256, S_LEN / 64);
    gemm_bf16_nt<<<grid, 256, 0, stream>>>(hs, Wqkv, qkv, S_LEN, QKV_COLS, HID);
  }
  // 2) RoPE on q and k
  {
    int total = S_LEN * (NQH + NKV) * 64;
    rope_kernel<<<(total + 255) / 256, 256, 0, stream>>>(qkv);
  }
  // 3) causal GQA flash attention -> attn (2048 x 4096)
  {
    dim3 grid(S_LEN / 64, NQH);
    attn_fa<<<grid, 128, 0, stream>>>(qkv, attn);
  }
  // 4) output projection: out[s,h] = sum_d attn[s,d] * Wo[h,d]
  {
    dim3 grid(HID / 256, S_LEN / 64);
    gemm_bf16_nt<<<grid, 256, 0, stream>>>(attn, Wo, outp, S_LEN, HID, HID);
  }
}
